// TreePoinTr_78864189489162
// MI455X (gfx1250) — matrix operand order repaired and checked by reading, not compiled, not run on hardware
//
#include <hip/hip_runtime.h>
#include <hip/hip_bf16.h>
#include <stdint.h>

// Collapsed TreePoinTr: attention over broadcast-KV degenerates to per-batch
// row vectors; only KNN/graph-conv (proxy) and the coarse rank-3 term are
// per-point. GEMMs are 32-row x K x N done with v_wmma_f32_16x16x32_bf16.

typedef __attribute__((ext_vector_type(16))) __bf16 v16bf;
typedef __attribute__((ext_vector_type(8)))  float  v8f;

__device__ __forceinline__ unsigned f32_mono(float f) {
  unsigned u = __float_as_uint(f);
  return (u & 0x80000000u) ? ~u : (u | 0x80000000u);
}
__device__ __forceinline__ float mono_f32(unsigned u) {
  return __uint_as_float((u & 0x80000000u) ? (u ^ 0x80000000u) : ~u);
}

// ---------------------------------------------------------------------------
// Kernel A: per-batch KNN(K=8) + 2 graph convs + max-pool -> proxy (32x128)
// One workgroup per batch, 512 threads (one per skeleton point).
// Skeleton coords staged to LDS with CDNA5 async-load; proxy max via ds atomics.
// ---------------------------------------------------------------------------
__global__ __launch_bounds__(512) void knn_gc_proxy_kernel(
    const float* __restrict__ skeleton,
    const float* __restrict__ w1r, const float* __restrict__ w1l, const float* __restrict__ b1,
    const float* __restrict__ w2r, const float* __restrict__ w2l, const float* __restrict__ b2,
    float* __restrict__ proxy)
{
  extern __shared__ float smem[];
  float*    sc   = smem;                       // 1536 floats: (512,3) coords
  float*    h1   = smem + 1536;                // 512*64 floats: gc1 output
  unsigned* pmax = (unsigned*)(smem + 1536 + 512 * 64); // 128 monotonic-uint maxes

  const int b = blockIdx.x;
  const int p = threadIdx.x;                   // point id 0..511
  const float* gsk = skeleton + (size_t)b * 1536;

  // Async copy of the 512x3 skeleton block into LDS (ASYNCcnt path).
#pragma unroll
  for (int r = 0; r < 3; ++r) {
    const int i = p + r * 512;
    const unsigned lds_off = (unsigned)(i * 4); // dynamic LDS base offset 0
    asm volatile("global_load_async_to_lds_b32 %0, %1, off"
                 :: "v"(lds_off), "v"(gsk + i) : "memory");
  }
  if (p < 128) pmax[p] = 0u;                   // below any mapped float
  asm volatile("s_wait_asynccnt 0x0" ::: "memory");
  __syncthreads();

  const float px = sc[3 * p + 0], py = sc[3 * p + 1], pz = sc[3 * p + 2];

  // --- KNN: keep 8 smallest squared distances (self excluded) -------------
  float bd[8];
  int   bi[8];
#pragma unroll
  for (int i = 0; i < 8; ++i) { bd[i] = 3.0e38f; bi[i] = 0; }
  for (int q = 0; q < 512; ++q) {
    const float dx = px - sc[3 * q + 0];
    const float dy = py - sc[3 * q + 1];
    const float dz = pz - sc[3 * q + 2];
    float d2 = fmaf(dx, dx, fmaf(dy, dy, dz * dz));
    if (q == p) d2 = 3.9e38f;
    if (d2 < bd[7]) {
      bd[7] = d2; bi[7] = q;
#pragma unroll
      for (int i = 7; i > 0; --i) {
        if (bd[i] < bd[i - 1]) {
          float td = bd[i]; bd[i] = bd[i - 1]; bd[i - 1] = td;
          int   ti = bi[i]; bi[i] = bi[i - 1]; bi[i - 1] = ti;
        }
      }
    }
  }

  // --- GC1: (512,3)->(512,64), relu ---------------------------------------
  float ax = 0.f, ay = 0.f, az = 0.f;
#pragma unroll
  for (int t = 0; t < 8; ++t) {
    const int q = bi[t];
    ax += sc[3 * q + 0]; ay += sc[3 * q + 1]; az += sc[3 * q + 2];
  }
  for (int j = 0; j < 64; ++j) {
    float h = b1[j];
    h = fmaf(px, w1r[j], h); h = fmaf(py, w1r[64 + j], h); h = fmaf(pz, w1r[128 + j], h);
    h = fmaf(ax, w1l[j], h); h = fmaf(ay, w1l[64 + j], h); h = fmaf(az, w1l[128 + j], h);
    h1[p * 64 + j] = h > 0.f ? h : 0.f;
  }
  __syncthreads();

  // --- GC2: (512,64)->(512,128), then block max-pool into pmax ------------
  float hp[64], ag[64];
  {
    const float4* hr = (const float4*)(h1 + p * 64);
#pragma unroll
    for (int k4 = 0; k4 < 16; ++k4) {
      float4 v = hr[k4];
      hp[4 * k4 + 0] = v.x; hp[4 * k4 + 1] = v.y; hp[4 * k4 + 2] = v.z; hp[4 * k4 + 3] = v.w;
      ag[4 * k4 + 0] = 0.f; ag[4 * k4 + 1] = 0.f; ag[4 * k4 + 2] = 0.f; ag[4 * k4 + 3] = 0.f;
    }
#pragma unroll
    for (int t = 0; t < 8; ++t) {
      const float4* nr = (const float4*)(h1 + bi[t] * 64);
#pragma unroll
      for (int k4 = 0; k4 < 16; ++k4) {
        float4 v = nr[k4];
        ag[4 * k4 + 0] += v.x; ag[4 * k4 + 1] += v.y; ag[4 * k4 + 2] += v.z; ag[4 * k4 + 3] += v.w;
      }
    }
  }
  for (int n = 0; n < 128; ++n) {
    float acc = b2[n];
#pragma unroll
    for (int k = 0; k < 64; ++k) {
      acc = fmaf(hp[k], w2r[k * 128 + n], acc);
      acc = fmaf(ag[k], w2l[k * 128 + n], acc);
    }
    atomicMax(&pmax[n], f32_mono(acc));        // ds_max_u32
  }
  __syncthreads();
  if (p < 128) proxy[b * 128 + p] = mono_f32(pmax[p]);
}

// ---------------------------------------------------------------------------
// GEMM: D(32xN) = epi(A(32xK) @ W + bias).  W element (k,n) = W[k*wsK + n*wsN]
// (wsN==C handles the `x @ W.T` layouts).  One wave per 16-wide N tile,
// two 16x16 M tiles via v_wmma_f32_16x16x32_bf16.  epilogue 1 = BN + leaky.
// ---------------------------------------------------------------------------
__global__ __launch_bounds__(256) void gemm32_bf16_wmma(
    const float* __restrict__ A, int lda, int K,
    const float* __restrict__ W, int wsK, int wsN,
    const float* __restrict__ bias,
    float* __restrict__ D, int ldo, int N,
    const float* __restrict__ bn_g, const float* __restrict__ bn_b,
    const float* __restrict__ bn_m, const float* __restrict__ bn_v,
    int epilogue)
{
  const int lane  = threadIdx.x & 31;
  const int wave  = threadIdx.x >> 5;
  const int ntile = blockIdx.x * 8 + wave;
  if (ntile * 16 >= N) return;                 // uniform per wave: EXEC stays full
  const int n0    = ntile * 16;
  const int nlane = lane & 15;
  const int half  = lane >> 4;

  v8f c0 = {}; v8f c1 = {};
  const float* arow0 = A + (size_t)nlane * lda;
  const float* arow1 = A + (size_t)(nlane + 16) * lda;

  for (int k0 = 0; k0 < K; k0 += 32) {
    v16bf a0, a1, bfrag;
    // A fragment (16x32 bf16): lane=M row; VGPR v -> K = 16*(v>=4)+8*half+2*(v&3)
#pragma unroll
    for (int v = 0; v < 8; ++v) {
      const int kb = k0 + ((v >> 2) << 4) + (half << 3) + ((v & 3) << 1);
      a0[2 * v]     = (__bf16)arow0[kb];
      a0[2 * v + 1] = (__bf16)arow0[kb + 1];
      a1[2 * v]     = (__bf16)arow1[kb];
      a1[2 * v + 1] = (__bf16)arow1[kb + 1];
    }
    // B fragment (32x16 bf16): lane = N column; VGPR v -> K = 16*(v>=4)+8*half+2*(v&3)
    const float* wrow = W + (size_t)k0 * wsK + (size_t)(n0 + nlane) * wsN;
#pragma unroll
    for (int v = 0; v < 8; ++v) {
      const int kr = ((v >> 2) << 4) + (half << 3) + ((v & 3) << 1);
      bfrag[2 * v]     = (__bf16)wrow[(size_t)kr * wsK];
      bfrag[2 * v + 1] = (__bf16)wrow[(size_t)(kr + 1) * wsK];
    }
    if (k0 + 32 < K)                           // global_prefetch_b8 next K tile
      __builtin_prefetch(W + (size_t)(k0 + 32 + lane) * wsK + (size_t)n0 * wsN, 0, 0);

    c0 = __builtin_amdgcn_wmma_f32_16x16x32_bf16(false, a0, false, bfrag, (short)0, c0, false, false);
    c1 = __builtin_amdgcn_wmma_f32_16x16x32_bf16(false, a1, false, bfrag, (short)0, c1, false, false);
  }

  const int n = n0 + nlane;
  const float bia = bias[n];
  float scale = 1.f, shift = 0.f;
  if (epilogue == 1) {
    const float inv = rsqrtf(bn_v[n] + 1e-5f);
    scale = bn_g[n] * inv;
    shift = bn_b[n] - bn_m[n] * scale;
  }
  // D layout: lane -> N; VGPR r -> M = r + 8*half (tile0), +16 (tile1)
#pragma unroll
  for (int r = 0; r < 8; ++r) {
    const int m = r + 8 * half;
    float x0 = c0[r] + bia;
    float x1 = c1[r] + bia;
    if (epilogue == 1) {
      x0 = x0 * scale + shift; x0 = x0 >= 0.f ? x0 : 0.2f * x0;
      x1 = x1 * scale + shift; x1 = x1 >= 0.f ? x1 : 0.2f * x1;
    }
    D[(size_t)m * ldo + n]        = x0;
    D[(size_t)(m + 16) * ldo + n] = x1;
  }
}

// ---------------------------------------------------------------------------
// Final: out[b,m,:] = base[b,:] + coarse[b,m,:] @ red_w[1408:1411,:]
// Pure HBM-bandwidth kernel (22 MB write).
// ---------------------------------------------------------------------------
__global__ __launch_bounds__(256) void final_out_kernel(
    const float* __restrict__ base, const float* __restrict__ coarse,
    const float* __restrict__ w3, float* __restrict__ out, int total)
{
  const int i = blockIdx.x * blockDim.x + threadIdx.x;
  if (i >= total) return;
  const int n   = i % 384;
  const int row = i / 384;            // b*448 + m
  const int b   = row / 448;
  const float c0 = coarse[row * 3 + 0];
  const float c1 = coarse[row * 3 + 1];
  const float c2 = coarse[row * 3 + 2];
  float v = base[b * 384 + n];
  v = fmaf(c0, w3[n], v);
  v = fmaf(c1, w3[384 + n], v);
  v = fmaf(c2, w3[768 + n], v);
  out[i] = v;
}

extern "C" void kernel_launch(void* const* d_in, const int* in_sizes, int n_in,
                              void* d_out, int out_size, void* d_ws, size_t ws_size,
                              hipStream_t stream) {
  (void)in_sizes; (void)n_in; (void)out_size; (void)ws_size;
  const float* coarse     = (const float*)d_in[1];
  const float* skeleton   = (const float*)d_in[2];
  const float* gc1_wroot  = (const float*)d_in[3];
  const float* gc1_wrel   = (const float*)d_in[4];
  const float* gc1_b      = (const float*)d_in[5];
  const float* gc2_wroot  = (const float*)d_in[6];
  const float* gc2_wrel   = (const float*)d_in[7];
  const float* gc2_b      = (const float*)d_in[8];
  const float* proj_w     = (const float*)d_in[9];
  const float* proj_b     = (const float*)d_in[10];
  const float* attn_in_w  = (const float*)d_in[11];
  const float* attn_in_b  = (const float*)d_in[12];
  const float* attn_out_w = (const float*)d_in[13];
  const float* attn_out_b = (const float*)d_in[14];
  const float* inc1_w     = (const float*)d_in[15];
  const float* inc1_b     = (const float*)d_in[16];
  const float* bn_g       = (const float*)d_in[17];
  const float* bn_b       = (const float*)d_in[18];
  const float* bn_m       = (const float*)d_in[19];
  const float* bn_v       = (const float*)d_in[20];
  const float* inc2_w     = (const float*)d_in[21];
  const float* inc2_b     = (const float*)d_in[22];
  const float* red_w      = (const float*)d_in[23];
  const float* red_b      = (const float*)d_in[24];
  float* out = (float*)d_out;

  float* ws    = (float*)d_ws;
  float* proxy = ws;                  // 32 x 128
  float* pvec  = proxy + 32 * 128;    // 32 x 384
  float* ctx   = pvec  + 32 * 384;    // 32 x 384
  float* cat   = ctx   + 32 * 384;    // 32 x 1408 : [g2 (1024) | qa (384)]
  float* g1b   = cat   + 32 * 1408;   // 32 x 1024
  float* base  = g1b   + 32 * 1024;   // 32 x 384

  const size_t shmem = (size_t)(1536 + 512 * 64 + 128) * sizeof(float);
  knn_gc_proxy_kernel<<<32, 512, shmem, stream>>>(
      skeleton, gc1_wroot, gc1_wrel, gc1_b, gc2_wroot, gc2_wrel, gc2_b, proxy);

  // pvec = proxy @ proj_w + proj_b                      (K=128, N=384)
  gemm32_bf16_wmma<<<3, 256, 0, stream>>>(proxy, 128, 128, proj_w, 384, 1,
      proj_b, pvec, 384, 384, nullptr, nullptr, nullptr, nullptr, 0);
  // ctx = pvec @ wv.T + bv   (wv = attn_in_w[2C:3C])    (K=384, N=384)
  gemm32_bf16_wmma<<<3, 256, 0, stream>>>(pvec, 384, 384, attn_in_w + 2 * 384 * 384, 1, 384,
      attn_in_b + 2 * 384, ctx, 384, 384, nullptr, nullptr, nullptr, nullptr, 0);
  // qa = ctx @ attn_out_w.T + b  -> cat columns 1024..1407
  gemm32_bf16_wmma<<<3, 256, 0, stream>>>(ctx, 384, 384, attn_out_w, 1, 384,
      attn_out_b, cat + 1024, 1408, 384, nullptr, nullptr, nullptr, nullptr, 0);
  // g1 = BN+leaky(qa @ inc1_w.T + b)                    (K=384, N=1024)
  gemm32_bf16_wmma<<<8, 256, 0, stream>>>(cat + 1024, 1408, 384, inc1_w, 1, 384,
      inc1_b, g1b, 1024, 1024, bn_g, bn_b, bn_m, bn_v, 1);
  // g2 = g1 @ inc2_w.T + b  -> cat columns 0..1023      (K=1024, N=1024)
  gemm32_bf16_wmma<<<8, 256, 0, stream>>>(g1b, 1024, 1024, inc2_w, 1, 1024,
      inc2_b, cat, 1408, 1024, nullptr, nullptr, nullptr, nullptr, 0);
  // base = [g2|qa] @ red_w[0:1408] + red_b              (K=1408, N=384)
  gemm32_bf16_wmma<<<3, 256, 0, stream>>>(cat, 1408, 1408, red_w, 384, 1,
      red_b, base, 384, 384, nullptr, nullptr, nullptr, nullptr, 0);

  const int total = 32 * 448 * 384;
  final_out_kernel<<<(total + 255) / 256, 256, 0, stream>>>(
      base, coarse, red_w + 1408 * 384, out, total);
}
